// GraphEncoder_57870389347019
// MI455X (gfx1250) — compile-verified, matrix operand order checked by reading, not run on hardware
//
#include <hip/hip_runtime.h>

// ---------------------------------------------------------------------------
// Types for CDNA5 WMMA (wave32)
// ---------------------------------------------------------------------------
typedef _Float16 h8   __attribute__((ext_vector_type(8)));
typedef _Float16 v16h __attribute__((ext_vector_type(16)));
typedef float    v8f  __attribute__((ext_vector_type(8)));

#define NNODES 4096
#define DMODEL 512
#define NHEADS 8
#define HIDC   64
#define DFFN   256

__device__ __forceinline__ v8f wmma32(v16h a, v16h b, v8f c) {
  // D = A(16x32 f16) * B(32x16 f16) + C(16x16 f32)
  return __builtin_amdgcn_wmma_f32_16x16x32_f16(false, a, false, b, (short)0, c,
                                                false, false);
}

__device__ __forceinline__ v8f zero8() {
  v8f z = {0.f, 0.f, 0.f, 0.f, 0.f, 0.f, 0.f, 0.f};
  return z;
}

// Load a 16x32 f16 fragment (A layout; B uses the same pattern when the
// matrix is stored contraction-contiguous per column, i.e. [N,K] row-major).
// Lane l (0..15): row = l, K = kbase+0..7 (lo) and kbase+16..23 (hi)
// Lane l (16..31): row = l-16, K = kbase+8..15 (lo) and kbase+24..31 (hi)
__device__ __forceinline__ v16h load_frag16(const _Float16* base, int stride,
                                            int lane, int kbase) {
  const int r  = lane & 15;
  const int ks = (lane & 16) ? 8 : 0;
  const _Float16* p = base + r * stride + kbase + ks;
  h8 lo = *(const h8*)(p);
  h8 hi = *(const h8*)(p + 16);
  v16h out;
#pragma unroll
  for (int i = 0; i < 8; ++i) { out[i] = lo[i]; out[i + 8] = hi[i]; }
  return out;
}

// ---------------------------------------------------------------------------
// f32 -> f16 conversion
// ---------------------------------------------------------------------------
__global__ __launch_bounds__(256) void cvt_f16_kernel(
    const float* __restrict__ in, _Float16* __restrict__ out, int n) {
  int i = blockIdx.x * 256 + threadIdx.x;
  if (i < n) out[i] = (_Float16)in[i];
}

// ---------------------------------------------------------------------------
// Generic WMMA GEMM: C[M,N] = A[M,K](f16) * B[N,K](f16)^T  (+bias, relu)
// Block: 256 threads = 8 waves, tile 64(M) x 128(N), K staged in 32-steps.
// Wave w: wm = w&1 (M half), wn = w>>1, arranged 2x4 -> 32x32 per wave.
// M % 64 == 0, N % 128 == 0, K % 32 == 0 required.
// ---------------------------------------------------------------------------
__global__ __launch_bounds__(256) void gemm_f16_kernel(
    const _Float16* __restrict__ A, const _Float16* __restrict__ B,
    const float* __restrict__ bias, float* __restrict__ Cf,
    _Float16* __restrict__ Ch, int M, int N, int K, int relu) {
  __shared__ __align__(16) _Float16 As[64 * 40];   // 64 rows x 32 (+8 pad)
  __shared__ __align__(16) _Float16 Bs[128 * 40];  // 128 rows x 32 (+8 pad)
  const int tid  = threadIdx.x;
  const int wave = tid >> 5, lane = tid & 31;
  const int wm = wave & 1, wn = wave >> 1;   // 2 x 4 wave grid
  const int row0 = blockIdx.y * 64, col0 = blockIdx.x * 128;

  v8f acc[2][2];
#pragma unroll
  for (int i = 0; i < 2; ++i)
#pragma unroll
    for (int j = 0; j < 2; ++j) acc[i][j] = zero8();

  const int ar = tid >> 2, ac = (tid & 3) * 8;  // A: 64 rows x 4 chunks
  const int br = tid >> 1, bc = (tid & 1) * 16; // B: 128 rows x 2 chunks(16)

  for (int k0 = 0; k0 < K; k0 += 32) {
    *(h8*)&As[ar * 40 + ac] =
        *(const h8*)(A + (size_t)(row0 + ar) * K + k0 + ac);
    *(h8*)&Bs[br * 40 + bc] =
        *(const h8*)(B + (size_t)(col0 + br) * K + k0 + bc);
    *(h8*)&Bs[br * 40 + bc + 8] =
        *(const h8*)(B + (size_t)(col0 + br) * K + k0 + bc + 8);
    __syncthreads();

    v16h a0 = load_frag16(&As[(wm * 32) * 40], 40, lane, 0);
    v16h a1 = load_frag16(&As[(wm * 32 + 16) * 40], 40, lane, 0);
    v16h b0 = load_frag16(&Bs[(wn * 32) * 40], 40, lane, 0);
    v16h b1 = load_frag16(&Bs[(wn * 32 + 16) * 40], 40, lane, 0);
    acc[0][0] = wmma32(a0, b0, acc[0][0]);
    acc[0][1] = wmma32(a0, b1, acc[0][1]);
    acc[1][0] = wmma32(a1, b0, acc[1][0]);
    acc[1][1] = wmma32(a1, b1, acc[1][1]);
    __syncthreads();
  }

  const int rr = (lane >> 4) * 8, cc = lane & 15;
#pragma unroll
  for (int mi = 0; mi < 2; ++mi)
#pragma unroll
    for (int ni = 0; ni < 2; ++ni) {
      const int gcol = col0 + wn * 32 + ni * 16 + cc;
      const float bb = bias ? bias[gcol] : 0.f;
#pragma unroll
      for (int r = 0; r < 8; ++r) {
        const int grow = row0 + wm * 32 + mi * 16 + r + rr;
        float v = acc[mi][ni][r] + bb;
        if (relu) v = fmaxf(v, 0.f);
        if (Cf) Cf[(size_t)grow * N + gcol] = v;
        if (Ch) Ch[(size_t)grow * N + gcol] = (_Float16)v;
      }
    }
}

// ---------------------------------------------------------------------------
// Flash attention: one block (4 waves, 128 thr) per (head, 64-query block).
// qkv is f16 [4096, 1536] (q | k | v, head-major 64-wide slices).
// Writes f16 attention output [4096, 512].
// ---------------------------------------------------------------------------
__global__ __launch_bounds__(128) void flash_attn_kernel(
    const _Float16* __restrict__ qkv, _Float16* __restrict__ attn_out) {
  __shared__ __align__(16) _Float16 Qs[64 * 72];
  __shared__ __align__(16) _Float16 Ks[64 * 72];
  __shared__ __align__(16) _Float16 Vt[64 * 72];  // transposed: [hid][key]
  __shared__ __align__(16) _Float16 Ps[64 * 72];
  const int head = blockIdx.x & 7;
  const int r0   = (blockIdx.x >> 3) << 6;
  const int tid  = threadIdx.x, wave = tid >> 5, lane = tid & 31;
  const int lr = tid >> 1, lc = (tid & 1) * 32;

  {  // load Q block [64 x 64]
    const _Float16* qg = qkv + (size_t)(r0 + lr) * 1536 + head * 64 + lc;
#pragma unroll
    for (int ch = 0; ch < 4; ++ch)
      *(h8*)&Qs[lr * 72 + lc + ch * 8] = *(const h8*)(qg + ch * 8);
  }
  __syncthreads();

  const int qr = wave * 16;
  const v16h aq0 = load_frag16(&Qs[qr * 72], 72, lane, 0);
  const v16h aq1 = load_frag16(&Qs[qr * 72], 72, lane, 32);

  v8f O[4];
#pragma unroll
  for (int j = 0; j < 4; ++j) O[j] = zero8();
  float rm[8], rl[8];
#pragma unroll
  for (int r = 0; r < 8; ++r) { rm[r] = -__builtin_inff(); rl[r] = 0.f; }

  for (int kb = 0; kb < 64; ++kb) {
    {  // stage K [key][hid] and V transposed [hid][key]
      const _Float16* kg =
          qkv + (size_t)(kb * 64 + lr) * 1536 + 512 + head * 64 + lc;
      const _Float16* vg = kg + 512;
#pragma unroll
      for (int ch = 0; ch < 4; ++ch)
        *(h8*)&Ks[lr * 72 + lc + ch * 8] = *(const h8*)(kg + ch * 8);
#pragma unroll
      for (int ch = 0; ch < 4; ++ch) {
        h8 vv = *(const h8*)(vg + ch * 8);
#pragma unroll
        for (int j = 0; j < 8; ++j) Vt[(lc + ch * 8 + j) * 72 + lr] = vv[j];
      }
    }
    __syncthreads();

    // S = Q * K^T  (scaled)
    v8f S[4];
#pragma unroll
    for (int j = 0; j < 4; ++j) {
      v16h b0 = load_frag16(&Ks[(j * 16) * 72], 72, lane, 0);
      v16h b1 = load_frag16(&Ks[(j * 16) * 72], 72, lane, 32);
      v8f s = zero8();
      s = wmma32(aq0, b0, s);
      s = wmma32(aq1, b1, s);
      S[j] = s;
    }
#pragma unroll
    for (int j = 0; j < 4; ++j)
#pragma unroll
      for (int r = 0; r < 8; ++r) S[j][r] *= 0.125f;  // 1/sqrt(64)

    // online softmax: row max / rescale
    float nm[8];
#pragma unroll
    for (int r = 0; r < 8; ++r) {
      float v = fmaxf(fmaxf(S[0][r], S[1][r]), fmaxf(S[2][r], S[3][r]));
      v = fmaxf(v, __shfl_xor(v, 1));
      v = fmaxf(v, __shfl_xor(v, 2));
      v = fmaxf(v, __shfl_xor(v, 4));
      v = fmaxf(v, __shfl_xor(v, 8));
      nm[r] = fmaxf(rm[r], v);
      const float sc = __expf(rm[r] - nm[r]);
      rm[r] = nm[r];
      rl[r] *= sc;
#pragma unroll
      for (int j = 0; j < 4; ++j) O[j][r] *= sc;
    }
#pragma unroll
    for (int j = 0; j < 4; ++j)
#pragma unroll
      for (int r = 0; r < 8; ++r) S[j][r] = __expf(S[j][r] - nm[r]);
#pragma unroll
    for (int r = 0; r < 8; ++r) {
      float s = (S[0][r] + S[1][r]) + (S[2][r] + S[3][r]);
      s += __shfl_xor(s, 1);
      s += __shfl_xor(s, 2);
      s += __shfl_xor(s, 4);
      s += __shfl_xor(s, 8);
      rl[r] += s;
    }

    // P (C-layout) -> LDS (row-major f16), wave-private rows
    const int rr = (lane >> 4) * 8, cc = lane & 15;
#pragma unroll
    for (int j = 0; j < 4; ++j)
#pragma unroll
      for (int r = 0; r < 8; ++r)
        Ps[(qr + r + rr) * 72 + j * 16 + cc] = (_Float16)S[j][r];

    // O += P * V
    v16h ap0 = load_frag16(&Ps[qr * 72], 72, lane, 0);
    v16h ap1 = load_frag16(&Ps[qr * 72], 72, lane, 32);
#pragma unroll
    for (int j = 0; j < 4; ++j) {
      v16h b0 = load_frag16(&Vt[(j * 16) * 72], 72, lane, 0);
      v16h b1 = load_frag16(&Vt[(j * 16) * 72], 72, lane, 32);
      O[j] = wmma32(ap0, b0, O[j]);
      O[j] = wmma32(ap1, b1, O[j]);
    }
    __syncthreads();
  }

  const int rr = (lane >> 4) * 8, cc = lane & 15;
#pragma unroll
  for (int r = 0; r < 8; ++r) rl[r] = 1.0f / rl[r];
#pragma unroll
  for (int j = 0; j < 4; ++j)
#pragma unroll
    for (int r = 0; r < 8; ++r)
      attn_out[(size_t)(r0 + qr + r + rr) * DMODEL + head * 64 + j * 16 + cc] =
          (_Float16)(O[j][r] * rl[r]);
}

// ---------------------------------------------------------------------------
// LayerNorm(a + b) * g + be ; one wave per 512-wide row.
// ---------------------------------------------------------------------------
__global__ __launch_bounds__(256) void ln_kernel(
    const float* __restrict__ a, const float* __restrict__ b,
    const float* __restrict__ g, const float* __restrict__ be,
    float* __restrict__ of, _Float16* __restrict__ oh) {
  const int wave = threadIdx.x >> 5, lane = threadIdx.x & 31;
  const int row = blockIdx.x * 8 + wave;
  const float* pa = a + (size_t)row * DMODEL;
  const float* pb = b + (size_t)row * DMODEL;
  float v[16];
  float s = 0.f;
#pragma unroll
  for (int k = 0; k < 4; ++k) {
    const float4 x4 = *(const float4*)(pa + lane * 4 + k * 128);
    const float4 y4 = *(const float4*)(pb + lane * 4 + k * 128);
    v[k * 4 + 0] = x4.x + y4.x; v[k * 4 + 1] = x4.y + y4.y;
    v[k * 4 + 2] = x4.z + y4.z; v[k * 4 + 3] = x4.w + y4.w;
    s += v[k * 4 + 0] + v[k * 4 + 1] + v[k * 4 + 2] + v[k * 4 + 3];
  }
#pragma unroll
  for (int m = 16; m >= 1; m >>= 1) s += __shfl_xor(s, m);
  const float mu = s * (1.f / 512.f);
  float q = 0.f;
#pragma unroll
  for (int i = 0; i < 16; ++i) { const float d = v[i] - mu; q += d * d; }
#pragma unroll
  for (int m = 16; m >= 1; m >>= 1) q += __shfl_xor(q, m);
  const float rstd = rsqrtf(q * (1.f / 512.f) + 1e-5f);
#pragma unroll
  for (int k = 0; k < 4; ++k)
#pragma unroll
    for (int j = 0; j < 4; ++j) {
      const int col = lane * 4 + k * 128 + j;
      const float y = (v[k * 4 + j] - mu) * rstd * g[col] + be[col];
      of[(size_t)row * DMODEL + col] = y;
      if (oh) oh[(size_t)row * DMODEL + col] = (_Float16)y;
    }
}

// ---------------------------------------------------------------------------
// GAT pieces
// ---------------------------------------------------------------------------
__device__ __forceinline__ int fenc(float f) {  // order-preserving float->int
  const int i = __float_as_int(f);
  return i >= 0 ? i : (i ^ 0x7fffffff);
}
__device__ __forceinline__ float fdec(int i) {
  return __int_as_float(i >= 0 ? i : (i ^ 0x7fffffff));
}

__global__ __launch_bounds__(256) void gat_scores_kernel(
    const float* __restrict__ h, const float* __restrict__ ws,
    const float* __restrict__ wd, float* __restrict__ a_src,
    float* __restrict__ a_dst) {
  const int t = blockIdx.x * 256 + threadIdx.x;  // NNODES * NHEADS threads
  const int n = t >> 3, hh = t & 7;
  const float* hp = h + (size_t)n * DMODEL + hh * HIDC;
  const float* ps = ws + hh * HIDC;
  const float* pd = wd + hh * HIDC;
  float s = 0.f, d = 0.f;
#pragma unroll 8
  for (int c = 0; c < HIDC; ++c) { s += hp[c] * ps[c]; d += hp[c] * pd[c]; }
  a_src[t] = s;
  a_dst[t] = d;
}

__global__ __launch_bounds__(256) void gat_init_kernel(int* __restrict__ emax,
                                                       float* __restrict__ den) {
  const int i = blockIdx.x * 256 + threadIdx.x;  // NNODES * NHEADS
  emax[i] = fenc(-3.4e38f);
  den[i] = 0.f;
}

__global__ __launch_bounds__(256) void bias_init_kernel(
    float* __restrict__ h, const float* __restrict__ bias) {
  const int i = blockIdx.x * 256 + threadIdx.x;  // NNODES * DMODEL
  h[i] = bias[i & (DMODEL - 1)];
}

__device__ __forceinline__ float edge_logit(const float* as, const float* ad,
                                            int s, int d, int hh) {
  const float v = as[s * 8 + hh] + ad[d * 8 + hh];
  return v > 0.f ? v : 0.2f * v;  // leaky_relu 0.2
}

__global__ __launch_bounds__(256) void edge_max_kernel(
    const int* __restrict__ ei, const float* __restrict__ as,
    const float* __restrict__ ad, int* __restrict__ emax, int E, int NE) {
  const int e = blockIdx.x * 256 + threadIdx.x;
  if (e >= NE) return;
  const int s = (e < E) ? ei[e] : (e - E);
  const int d = (e < E) ? ei[E + e] : (e - E);
#pragma unroll
  for (int hh = 0; hh < 8; ++hh)
    atomicMax(&emax[d * 8 + hh], fenc(edge_logit(as, ad, s, d, hh)));
}

__global__ __launch_bounds__(256) void edge_exp_kernel(
    const int* __restrict__ ei, const float* __restrict__ as,
    const float* __restrict__ ad, const int* __restrict__ emax,
    float* __restrict__ p, float* __restrict__ den, int E, int NE) {
  const int e = blockIdx.x * 256 + threadIdx.x;
  if (e >= NE) return;
  const int s = (e < E) ? ei[e] : (e - E);
  const int d = (e < E) ? ei[E + e] : (e - E);
#pragma unroll
  for (int hh = 0; hh < 8; ++hh) {
    const float v = edge_logit(as, ad, s, d, hh);
    const float pv = expf(v - fdec(emax[d * 8 + hh]));
    p[(size_t)e * 8 + hh] = pv;
    atomicAdd(&den[d * 8 + hh], pv);
  }
}

__global__ __launch_bounds__(256) void gat_scatter_kernel(
    const int* __restrict__ ei, const float* __restrict__ p,
    const float* __restrict__ den, const float* __restrict__ hG,
    float* __restrict__ out, int E, int NE) {
  const int t = blockIdx.x * 256 + threadIdx.x;  // NE * DMODEL threads
  const int e = t >> 9;
  if (e >= NE) return;
  const int idx = t & (DMODEL - 1);
  const int hh = idx >> 6;
  const int s = (e < E) ? ei[e] : (e - E);
  const int d = (e < E) ? ei[E + e] : (e - E);
  const float alpha = p[(size_t)e * 8 + hh] / (den[d * 8 + hh] + 1e-16f);
  atomicAdd(&out[(size_t)d * DMODEL + idx],
            alpha * hG[(size_t)s * DMODEL + idx]);
}

// ---------------------------------------------------------------------------
// Launch
// ---------------------------------------------------------------------------
extern "C" void kernel_launch(void* const* d_in, const int* in_sizes, int n_in,
                              void* d_out, int out_size, void* d_ws,
                              size_t ws_size, hipStream_t stream) {
  const float* x          = (const float*)d_in[0];
  const int*   edge_index = (const int*)d_in[1];
  const float* gat_lin_w  = (const float*)d_in[3];
  const float* att_src    = (const float*)d_in[4];
  const float* att_dst    = (const float*)d_in[5];
  const float* gat_bias   = (const float*)d_in[6];
  const float* qkv_w      = (const float*)d_in[7];
  const float* qkv_b      = (const float*)d_in[8];
  const float* out_w      = (const float*)d_in[9];
  const float* out_b      = (const float*)d_in[10];
  const float* ln1_g      = (const float*)d_in[11];
  const float* ln1_b      = (const float*)d_in[12];
  const float* ff1_w      = (const float*)d_in[13];
  const float* ff1_b      = (const float*)d_in[14];
  const float* ff2_w      = (const float*)d_in[15];
  const float* ff2_b      = (const float*)d_in[16];
  const float* ln2_g      = (const float*)d_in[17];
  const float* ln2_b      = (const float*)d_in[18];

  const int E  = in_sizes[1] / 2;
  const int NE = E + NNODES;

  // workspace bump allocator (256B aligned)
  char* base = (char*)d_ws;
  size_t off = 0;
  auto alloc = [&](size_t bytes) -> void* {
    void* p = base + off;
    off = (off + bytes + 255) & ~(size_t)255;
    return p;
  };
  _Float16* xh      = (_Float16*)alloc((size_t)NNODES * 128 * 2);
  _Float16* gatwh   = (_Float16*)alloc((size_t)DMODEL * 128 * 2);
  _Float16* qkvwh   = (_Float16*)alloc((size_t)2 * 3 * DMODEL * DMODEL * 2);
  _Float16* outwh   = (_Float16*)alloc((size_t)2 * DMODEL * DMODEL * 2);
  _Float16* ff1wh   = (_Float16*)alloc((size_t)2 * DFFN * DMODEL * 2);
  _Float16* ff2wh   = (_Float16*)alloc((size_t)2 * DMODEL * DFFN * 2);
  float*    hG      = (float*)alloc((size_t)NNODES * DMODEL * 4);
  float*    a_src   = (float*)alloc((size_t)NNODES * NHEADS * 4);
  float*    a_dst   = (float*)alloc((size_t)NNODES * NHEADS * 4);
  int*      emax    = (int*)alloc((size_t)NNODES * NHEADS * 4);
  float*    denom   = (float*)alloc((size_t)NNODES * NHEADS * 4);
  float*    pbuf    = (float*)alloc((size_t)NE * NHEADS * 4);
  float*    h       = (float*)alloc((size_t)NNODES * DMODEL * 4);
  _Float16* h_f16   = (_Float16*)alloc((size_t)NNODES * DMODEL * 2);
  _Float16* qkv_f16 = (_Float16*)alloc((size_t)NNODES * 3 * DMODEL * 2);
  _Float16* attn16  = (_Float16*)alloc((size_t)NNODES * DMODEL * 2);
  float*    proj    = (float*)alloc((size_t)NNODES * DMODEL * 4);
  float*    h2      = (float*)alloc((size_t)NNODES * DMODEL * 4);
  _Float16* h2_f16  = (_Float16*)alloc((size_t)NNODES * DMODEL * 2);
  _Float16* ffmid16 = (_Float16*)alloc((size_t)NNODES * DFFN * 2);
  float*    ffout   = (float*)alloc((size_t)NNODES * DMODEL * 4);

  auto cvt = [&](const float* src, _Float16* dst, int n) {
    cvt_f16_kernel<<<dim3((n + 255) / 256), dim3(256), 0, stream>>>(src, dst, n);
  };
  auto gemm = [&](const _Float16* A, const _Float16* Bw, const float* bias,
                  float* Cf, _Float16* Ch, int M, int N, int K, int relu) {
    gemm_f16_kernel<<<dim3(N / 128, M / 64), dim3(256), 0, stream>>>(
        A, Bw, bias, Cf, Ch, M, N, K, relu);
  };

  // ---- convert operands to f16 ----
  cvt(x, xh, NNODES * 128);
  cvt(gat_lin_w, gatwh, DMODEL * 128);
  cvt(qkv_w, qkvwh, 2 * 3 * DMODEL * DMODEL);
  cvt(out_w, outwh, 2 * DMODEL * DMODEL);
  cvt(ff1_w, ff1wh, 2 * DFFN * DMODEL);
  cvt(ff2_w, ff2wh, 2 * DMODEL * DFFN);

  // ---- GAT ----
  gemm(xh, gatwh, nullptr, hG, nullptr, NNODES, DMODEL, 128, 0);
  gat_scores_kernel<<<dim3(NNODES * NHEADS / 256), dim3(256), 0, stream>>>(
      hG, att_src, att_dst, a_src, a_dst);
  gat_init_kernel<<<dim3(NNODES * NHEADS / 256), dim3(256), 0, stream>>>(emax,
                                                                         denom);
  bias_init_kernel<<<dim3(NNODES * DMODEL / 256), dim3(256), 0, stream>>>(
      h, gat_bias);
  const int egrid = (NE + 255) / 256;
  edge_max_kernel<<<dim3(egrid), dim3(256), 0, stream>>>(edge_index, a_src,
                                                         a_dst, emax, E, NE);
  edge_exp_kernel<<<dim3(egrid), dim3(256), 0, stream>>>(
      edge_index, a_src, a_dst, emax, pbuf, denom, E, NE);
  gat_scatter_kernel<<<dim3(NE * (DMODEL / 256)), dim3(256), 0, stream>>>(
      edge_index, pbuf, denom, hG, h, E, NE);
  cvt(h, h_f16, NNODES * DMODEL);

  // ---- 2 transformer encoder layers (post-norm) ----
  for (int i = 0; i < 2; ++i) {
    gemm(h_f16, qkvwh + (size_t)i * 3 * DMODEL * DMODEL,
         qkv_b + (size_t)i * 3 * DMODEL, nullptr, qkv_f16, NNODES, 3 * DMODEL,
         DMODEL, 0);
    flash_attn_kernel<<<dim3(NHEADS * (NNODES / 64)), dim3(128), 0, stream>>>(
        qkv_f16, attn16);
    gemm(attn16, outwh + (size_t)i * DMODEL * DMODEL,
         out_b + (size_t)i * DMODEL, proj, nullptr, NNODES, DMODEL, DMODEL, 0);
    ln_kernel<<<dim3(NNODES / 8), dim3(256), 0, stream>>>(
        h, proj, ln1_g + (size_t)i * DMODEL, ln1_b + (size_t)i * DMODEL, h2,
        h2_f16);
    gemm(h2_f16, ff1wh + (size_t)i * DFFN * DMODEL, ff1_b + (size_t)i * DFFN,
         nullptr, ffmid16, NNODES, DFFN, DMODEL, 1);
    gemm(ffmid16, ff2wh + (size_t)i * DMODEL * DFFN,
         ff2_b + (size_t)i * DMODEL, ffout, nullptr, NNODES, DMODEL, DFFN, 0);
    const bool last = (i == 1);
    ln_kernel<<<dim3(NNODES / 8), dim3(256), 0, stream>>>(
        h2, ffout, ln2_g + (size_t)i * DMODEL, ln2_b + (size_t)i * DMODEL,
        last ? (float*)d_out : h, last ? nullptr : h_f16);
  }
}
